// StackedSelfAttentions_64252710748655
// MI455X (gfx1250) — matrix operand order repaired and checked by reading, compiled, not run on hardware
//
#include <hip/hip_runtime.h>

// ---------------------------------------------------------------------------
// Types for CDNA5 WMMA (wave32): v16bf A/B fragments, v8f C/D accumulators.
// ---------------------------------------------------------------------------
typedef __attribute__((ext_vector_type(16))) __bf16 v16bf;
typedef __attribute__((ext_vector_type(8)))  float  v8f;

union ABFrag {
    v16bf v;
    uint4 u[2];
};

__device__ __forceinline__ __bf16 f2bf(float f) {
    // round-to-nearest-even fp32 -> bf16
    unsigned u = __builtin_bit_cast(unsigned, f);
    u += 0x7FFFu + ((u >> 16) & 1u);
    unsigned short h = (unsigned short)(u >> 16);
    return __builtin_bit_cast(__bf16, h);
}

// pack two bf16 (low halves of a,b) -> one dword {b,a}
__device__ __forceinline__ unsigned pk_lo(unsigned a, unsigned b) {
    return (a & 0xFFFFu) | (b << 16);
}
__device__ __forceinline__ unsigned pk_hi(unsigned a, unsigned b) {
    return (a >> 16) | (b & 0xFFFF0000u);
}

// ---------------------------------------------------------------------------
// Generic bf16 GEMM: C[f32 and/or bf16] = A(MxK) @ B(KxN), per-head strides.
// Block tile 128x64, K-step 32. 256 threads = 8 waves (4x2 wave grid, each
// wave a 32x32 strip = 2x2 WMMA tiles). Software-pipelined: next tile is
// preloaded into registers while WMMAs consume the current LDS buffer
// (double-buffered LDS, one barrier per K-step). B staged transposed [n][k]
// with two K-rows packed per ds_store_b32 so B-fragments are two 16-byte
// LDS reads per lane.
// ---------------------------------------------------------------------------
#define BM 128
#define BN 64
#define BK 32

__global__ __launch_bounds__(256) void gemm_bf16_wmma(
    const __bf16* __restrict__ A, long aHead,
    const __bf16* __restrict__ B, long bHead,
    float* __restrict__ Cf, __bf16* __restrict__ Cb, long cHead,
    int K, int lda, int ldb, int ldc)
{
    __shared__ alignas(16) __bf16 As[2][BM][BK];
    __shared__ alignas(16) __bf16 Bs[2][BN][BK];   // transposed: Bs[n][k]

    const int h = blockIdx.z;
    A += (long)h * aHead;
    B += (long)h * bHead;
    const long coff = (long)h * cHead;

    const int m0   = blockIdx.y * BM;
    const int n0   = blockIdx.x * BN;
    const int tid  = threadIdx.x;
    const int wave = tid >> 5;
    const int lane = tid & 31;
    const int hl   = lane >> 4;       // half-wave select
    const int ln   = lane & 15;
    const int wm   = (wave >> 1) * 32;
    const int wn   = (wave & 1) * 32;

    // staging geometry
    const int ar = tid >> 1, ac = (tid & 1) * 16;        // A: 32B per thread
    const int bn = (tid & 15) * 4, bk = (tid >> 4) * 2;  // B: 2k x 4n per thread

    v8f acc[2][2] = {};

    // preload tile 0
    uint4 a0 = *(const uint4*)(A + (long)(m0 + ar) * lda + ac);
    uint4 a1 = *(const uint4*)(A + (long)(m0 + ar) * lda + ac + 8);
    uint2 b0 = *(const uint2*)(B + (long)bk * ldb + n0 + bn);
    uint2 b1 = *(const uint2*)(B + (long)(bk + 1) * ldb + n0 + bn);

    for (int k0 = 0; k0 < K; k0 += BK) {
        const int buf = (k0 >> 5) & 1;

        // commit staged registers to LDS[buf]
        *(uint4*)(&As[buf][ar][ac])     = a0;
        *(uint4*)(&As[buf][ar][ac + 8]) = a1;
        *(unsigned*)(&Bs[buf][bn + 0][bk]) = pk_lo(b0.x, b1.x);
        *(unsigned*)(&Bs[buf][bn + 1][bk]) = pk_hi(b0.x, b1.x);
        *(unsigned*)(&Bs[buf][bn + 2][bk]) = pk_lo(b0.y, b1.y);
        *(unsigned*)(&Bs[buf][bn + 3][bk]) = pk_hi(b0.y, b1.y);
        __syncthreads();

        // preload next tile (loads overlap the WMMAs below)
        if (k0 + BK < K) {
            a0 = *(const uint4*)(A + (long)(m0 + ar) * lda + k0 + BK + ac);
            a1 = *(const uint4*)(A + (long)(m0 + ar) * lda + k0 + BK + ac + 8);
            b0 = *(const uint2*)(B + (long)(k0 + BK + bk) * ldb + n0 + bn);
            b1 = *(const uint2*)(B + (long)(k0 + BK + bk + 1) * ldb + n0 + bn);
            if (k0 + 2 * BK < K)  // L2 prefetch of the tile after next
                __builtin_prefetch(A + (long)(m0 + ar) * lda + k0 + 2 * BK + ac, 0, 1);
        }

#pragma unroll
        for (int i = 0; i < 2; ++i) {
            ABFrag a;
            a.u[0] = *(const uint4*)(&As[buf][wm + i * 16 + ln][hl * 8]);
            a.u[1] = *(const uint4*)(&As[buf][wm + i * 16 + ln][16 + hl * 8]);
#pragma unroll
            for (int j = 0; j < 2; ++j) {
                ABFrag b;
                b.u[0] = *(const uint4*)(&Bs[buf][wn + j * 16 + ln][hl * 8]);
                b.u[1] = *(const uint4*)(&Bs[buf][wn + j * 16 + ln][16 + hl * 8]);
                acc[i][j] = __builtin_amdgcn_wmma_f32_16x16x32_bf16(
                    false, a.v, false, b.v, (short)0, acc[i][j], false, false);
            }
        }
        // single barrier per iteration: a wave can be at most one barrier
        // ahead, which touches only the other LDS buffer.
    }

    // --- store C: lane = column, VGPR index r = row within 16-row tile ---
#pragma unroll
    for (int i = 0; i < 2; ++i)
#pragma unroll
        for (int j = 0; j < 2; ++j)
#pragma unroll
            for (int r = 0; r < 8; ++r) {
                const int row = m0 + wm + i * 16 + hl * 8 + r;
                const int col = n0 + wn + j * 16 + ln;
                const float val = acc[i][j][r];
                if (Cf) Cf[coff + (long)row * ldc + col] = val;
                if (Cb) Cb[coff + (long)row * ldc + col] = f2bf(val);
            }
}

// ---------------------------------------------------------------------------
// Flash-attention: grid (S/128, B, H), 256 threads = 8 waves; each wave owns
// 16 query rows with a 16x256 f32 accumulator strip (16 x v8f). Key blocks
// of 32, software-pipelined with double-buffered K/V tiles. V staged
// transposed [vcol][key] with two key-rows packed per ds_store_b32. P makes
// an LDS round trip (f32 C-layout -> bf16 A-layout) guarded by s_wait_dscnt.
// ---------------------------------------------------------------------------
__global__ __launch_bounds__(256) void attn_wmma(
    const __bf16* __restrict__ Q,   // [H][B*S][64]
    const __bf16* __restrict__ Km,  // [H][B*S][64]
    const __bf16* __restrict__ Vm,  // [H][B*S][256]
    float* __restrict__ O)          // [H][B*S][256]
{
    __shared__ alignas(16) __bf16 Ks[2][32][64];    // [key][kd]  (B-frag for QK^T)
    __shared__ alignas(16) __bf16 VsT[2][256][32];  // [vcol][key] (B-frag for PV)
    __shared__ alignas(16) __bf16 Ps[8][16][32];    // per-wave P scratch

    const int h  = blockIdx.z;
    const int b  = blockIdx.y;
    const int qb = blockIdx.x;
    const int tid  = threadIdx.x;
    const int wave = tid >> 5;
    const int lane = tid & 31;
    const int hl   = lane >> 4;
    const int ln   = lane & 15;
    const int qr0  = qb * 128 + wave * 16;

    const __bf16* Qh = Q  + ((long)h * 4096 + b * 1024) * 64;
    const __bf16* Kh = Km + ((long)h * 4096 + b * 1024) * 64;
    const __bf16* Vh = Vm + ((long)h * 4096 + b * 1024) * 256;
    float*        Oh = O  + ((long)h * 4096 + b * 1024) * 256;

    // Q fragments for this wave's 16 rows: K-dim 0..31 and 32..63.
    ABFrag qf[2];
    {
        const __bf16* qrow = Qh + (long)(qr0 + ln) * 64;
#pragma unroll
        for (int f = 0; f < 2; ++f) {
            qf[f].u[0] = *(const uint4*)(qrow + f * 32 + hl * 8);
            qf[f].u[1] = *(const uint4*)(qrow + f * 32 + 16 + hl * 8);
        }
    }

    v8f o[16] = {};
    float rm[8], rl[8];
#pragma unroll
    for (int r = 0; r < 8; ++r) { rm[r] = -1e30f; rl[r] = 0.f; }

    // staging geometry
    const int skr = tid >> 3,        skc = (tid & 7) * 8;   // K tile: 16B/thread
    const int vkr = (tid >> 4) * 2,  vn  = (tid & 15) * 16; // V tile: 2 keys x 16 cols

    // preload key block 0
    uint4 kreg = *(const uint4*)(Kh + (long)skr * 64 + skc);
    uint4 v0a, v0b, v1a, v1b;
    {
        const __bf16* vp = Vh + (long)vkr * 256 + vn;
        v0a = *(const uint4*)(vp);       v0b = *(const uint4*)(vp + 8);
        v1a = *(const uint4*)(vp + 256); v1b = *(const uint4*)(vp + 264);
    }

    for (int kb = 0; kb < 1024; kb += 32) {
        const int buf = (kb >> 5) & 1;

        // commit staged K/V to LDS[buf]
        *(uint4*)(&Ks[buf][skr][skc]) = kreg;
        {
            const unsigned lo[8] = {v0a.x, v0a.y, v0a.z, v0a.w, v0b.x, v0b.y, v0b.z, v0b.w};
            const unsigned hi[8] = {v1a.x, v1a.y, v1a.z, v1a.w, v1b.x, v1b.y, v1b.z, v1b.w};
#pragma unroll
            for (int j = 0; j < 8; ++j) {
                *(unsigned*)(&VsT[buf][vn + 2 * j][vkr])     = pk_lo(lo[j], hi[j]);
                *(unsigned*)(&VsT[buf][vn + 2 * j + 1][vkr]) = pk_hi(lo[j], hi[j]);
            }
        }
        __syncthreads();

        // preload next key block (overlaps with compute below)
        if (kb + 32 < 1024) {
            kreg = *(const uint4*)(Kh + (long)(kb + 32 + skr) * 64 + skc);
            const __bf16* vp = Vh + (long)(kb + 32 + vkr) * 256 + vn;
            v0a = *(const uint4*)(vp);       v0b = *(const uint4*)(vp + 8);
            v1a = *(const uint4*)(vp + 256); v1b = *(const uint4*)(vp + 264);
        }

        // scores: two 16x16 tiles covering 32 keys, K-dim 64 = 2 WMMAs each
        v8f st[2];
#pragma unroll
        for (int jk = 0; jk < 2; ++jk) {
            v8f s = {};
#pragma unroll
            for (int f = 0; f < 2; ++f) {
                ABFrag bf;
                bf.u[0] = *(const uint4*)(&Ks[buf][jk * 16 + ln][f * 32 + hl * 8]);
                bf.u[1] = *(const uint4*)(&Ks[buf][jk * 16 + ln][f * 32 + 16 + hl * 8]);
                s = __builtin_amdgcn_wmma_f32_16x16x32_bf16(
                    false, qf[f].v, false, bf.v, (short)0, s, false, false);
            }
            st[jk] = s;
        }
#pragma unroll
        for (int jk = 0; jk < 2; ++jk)
#pragma unroll
            for (int r = 0; r < 8; ++r) st[jk][r] *= 0.125f;  // KD^-0.5

        // online softmax (rows live at fixed VGPR, 16 values across half-wave)
#pragma unroll
        for (int r = 0; r < 8; ++r) {
            float mx = fmaxf(st[0][r], st[1][r]);
#pragma unroll
            for (int off = 8; off; off >>= 1) mx = fmaxf(mx, __shfl_xor(mx, off, 32));
            const float nm   = fmaxf(rm[r], mx);
            const float corr = __expf(rm[r] - nm);
            const float p0 = __expf(st[0][r] - nm);
            const float p1 = __expf(st[1][r] - nm);
            st[0][r] = p0;
            st[1][r] = p1;
            float rs = p0 + p1;
#pragma unroll
            for (int off = 8; off; off >>= 1) rs += __shfl_xor(rs, off, 32);
            rl[r] = rl[r] * corr + rs;
            rm[r] = nm;
#pragma unroll
            for (int jn = 0; jn < 16; ++jn) o[jn][r] *= corr;
        }

        // P: C-layout f32 -> bf16 A-layout via per-wave LDS round trip
#pragma unroll
        for (int jk = 0; jk < 2; ++jk)
#pragma unroll
            for (int r = 0; r < 8; ++r)
                Ps[wave][hl * 8 + r][jk * 16 + ln] = f2bf(st[jk][r]);
        asm volatile("s_wait_dscnt 0" ::: "memory");
        ABFrag pf;
        pf.u[0] = *(const uint4*)(&Ps[wave][ln][hl * 8]);
        pf.u[1] = *(const uint4*)(&Ps[wave][ln][16 + hl * 8]);

        // O += P(16x32) @ V(32x256): 16 WMMAs across the 256-wide output
#pragma unroll
        for (int jn = 0; jn < 16; ++jn) {
            ABFrag bf;
            bf.u[0] = *(const uint4*)(&VsT[buf][jn * 16 + ln][hl * 8]);
            bf.u[1] = *(const uint4*)(&VsT[buf][jn * 16 + ln][16 + hl * 8]);
            o[jn] = __builtin_amdgcn_wmma_f32_16x16x32_bf16(
                false, pf.v, false, bf.v, (short)0, o[jn], false, false);
        }
        // single barrier per iteration (double-buffered tiles)
    }

    // epilogue: divide by row sum, store f32
#pragma unroll
    for (int r = 0; r < 8; ++r) {
        const float inv = 1.0f / rl[r];
        const int row = qr0 + hl * 8 + r;
#pragma unroll
        for (int jn = 0; jn < 16; ++jn)
            Oh[(long)row * 256 + jn * 16 + ln] = o[jn][r] * inv;
    }
}

// ---------------------------------------------------------------------------
// Residual add (+optional ReLU on the second operand), write y in place,
// accumulate per-(head,channel) sum / sumsq via global f32 atomics.
// ---------------------------------------------------------------------------
__global__ __launch_bounds__(256) void add_stats(
    float* __restrict__ net, const float* __restrict__ tmp,
    float* __restrict__ stats, int relu)
{
    const int h = blockIdx.y, chunk = blockIdx.x, c = threadIdx.x;
    const long base = (long)h * 4096 * 256 + (long)chunk * 128 * 256;
    float s = 0.f, s2 = 0.f;
    for (int r = 0; r < 128; ++r) {
        const long idx = base + (long)r * 256 + c;
        float bv = tmp[idx];
        if (relu) bv = fmaxf(bv, 0.f);
        const float y = net[idx] + bv;
        net[idx] = y;
        s += y;
        s2 += y * y;
    }
    atomicAdd(&stats[(h * 2 + 0) * 256 + c], s);
    atomicAdd(&stats[(h * 2 + 1) * 256 + c], s2);
}

__global__ void zero_stats(float* stats) {
    const int i = blockIdx.x * blockDim.x + threadIdx.x;
    if (i < 8 * 2 * 256) stats[i] = 0.f;
}

// BN normalize in place + emit bf16 copy for the next GEMM.
__global__ __launch_bounds__(256) void bn_norm(
    float* __restrict__ net, __bf16* __restrict__ netb,
    const float* __restrict__ stats,
    const float* __restrict__ gamma, const float* __restrict__ beta, int t)
{
    const int h = blockIdx.y, chunk = blockIdx.x, c = threadIdx.x;
    const float mean = stats[(h * 2 + 0) * 256 + c] * (1.f / 4096.f);
    const float var  = stats[(h * 2 + 1) * 256 + c] * (1.f / 4096.f) - mean * mean;
    const float inv  = rsqrtf(var + 1e-3f);
    const float g    = gamma[(h * 4 + t) * 256 + c];
    const float bt   = beta [(h * 4 + t) * 256 + c];
    const long base  = (long)h * 4096 * 256 + (long)chunk * 128 * 256;
    for (int r = 0; r < 128; ++r) {
        const long idx = base + (long)r * 256 + c;
        const float y = (net[idx] - mean) * inv * g + bt;
        net[idx]  = y;
        netb[idx] = f2bf(y);
    }
}

__global__ void f32_to_bf16(const float* __restrict__ in,
                            __bf16* __restrict__ out, long n)
{
    long i = (long)blockIdx.x * blockDim.x + threadIdx.x;
    const long stride = (long)gridDim.x * blockDim.x;
    for (; i < n; i += stride) out[i] = f2bf(in[i]);
}

// out[b][s][h][v] = net[h][b*1024+s][v]
__global__ void write_out(const float* __restrict__ net, float* __restrict__ out) {
    const int row = blockIdx.x;  // b*1024+s
    const int h   = blockIdx.y;
    const int v   = threadIdx.x;
    out[((long)row * 8 + h) * 256 + v] = net[((long)h * 4096 + row) * 256 + v];
}

// ---------------------------------------------------------------------------
// Host orchestration
// ---------------------------------------------------------------------------
extern "C" void kernel_launch(void* const* d_in, const int* in_sizes, int n_in,
                              void* d_out, int out_size, void* d_ws, size_t ws_size,
                              hipStream_t stream) {
    const float* x      = (const float*)d_in[0];
    const float* W_in   = (const float*)d_in[1];
    const float* Wq     = (const float*)d_in[2];
    const float* Wk     = (const float*)d_in[3];
    const float* Wv     = (const float*)d_in[4];
    const float* Wd     = (const float*)d_in[5];
    const float* gamma1 = (const float*)d_in[6];
    const float* beta1  = (const float*)d_in[7];
    const float* gamma2 = (const float*)d_in[8];
    const float* beta2  = (const float*)d_in[9];

    char* ws = (char*)d_ws;
    auto alloc = [&](size_t bytes) {
        char* p = ws;
        ws += (bytes + 255) & ~(size_t)255;
        return p;
    };
    float*  net_f = (float*) alloc(8L * 4096 * 256 * 4);
    __bf16* net_b = (__bf16*)alloc(8L * 4096 * 256 * 2);
    __bf16* q_b   = (__bf16*)alloc(8L * 4096 * 64 * 2);
    __bf16* k_b   = (__bf16*)alloc(8L * 4096 * 64 * 2);
    __bf16* v_b   = (__bf16*)alloc(8L * 4096 * 256 * 2);
    float*  tmp   = (float*) alloc(8L * 4096 * 256 * 4);
    float*  stats = (float*) alloc(8L * 2 * 256 * 4);
    __bf16* x_b   = (__bf16*)alloc(4096L * 256 * 2);
    __bf16* Win_b = (__bf16*)alloc(256L * 256 * 2);
    __bf16* Wq_b  = (__bf16*)alloc(8L * 4 * 256 * 64 * 2);
    __bf16* Wk_b  = (__bf16*)alloc(8L * 4 * 256 * 64 * 2);
    __bf16* Wv_b  = (__bf16*)alloc(8L * 4 * 256 * 256 * 2);
    __bf16* Wd_b  = (__bf16*)alloc(8L * 4 * 256 * 256 * 2);

    // fp32 -> bf16 conversions
    f32_to_bf16<<<512, 256, 0, stream>>>(x,    x_b,   4096L * 256);
    f32_to_bf16<<<128, 256, 0, stream>>>(W_in, Win_b, 256L * 256);
    f32_to_bf16<<<512, 256, 0, stream>>>(Wq,   Wq_b,  8L * 4 * 256 * 64);
    f32_to_bf16<<<512, 256, 0, stream>>>(Wk,   Wk_b,  8L * 4 * 256 * 64);
    f32_to_bf16<<<512, 256, 0, stream>>>(Wv,   Wv_b,  8L * 4 * 256 * 256);
    f32_to_bf16<<<512, 256, 0, stream>>>(Wd,   Wd_b,  8L * 4 * 256 * 256);

    const long NH = 4096L * 256;   // net head stride
    const long QH = 4096L * 64;

    // net0 = x @ W_in, replicated to all 8 heads (grid.z, strides 0 on A/B)
    gemm_bf16_wmma<<<dim3(4, 32, 8), 256, 0, stream>>>(
        x_b, 0, Win_b, 0, net_f, net_b, NH, 256, 256, 256, 256);

    for (int t = 0; t < 4; ++t) {
        // q / k / v projections
        gemm_bf16_wmma<<<dim3(1, 32, 8), 256, 0, stream>>>(
            net_b, NH, Wq_b + (long)t * 256 * 64, 4L * 256 * 64,
            nullptr, q_b, QH, 256, 256, 64, 64);
        gemm_bf16_wmma<<<dim3(1, 32, 8), 256, 0, stream>>>(
            net_b, NH, Wk_b + (long)t * 256 * 64, 4L * 256 * 64,
            nullptr, k_b, QH, 256, 256, 64, 64);
        gemm_bf16_wmma<<<dim3(4, 32, 8), 256, 0, stream>>>(
            net_b, NH, Wv_b + (long)t * 256 * 256, 4L * 256 * 256,
            nullptr, v_b, NH, 256, 256, 256, 256);

        // attention -> tmp (f32)
        attn_wmma<<<dim3(8, 4, 8), 256, 0, stream>>>(q_b, k_b, v_b, tmp);

        // net = BN(net + attn_out)
        zero_stats<<<16, 256, 0, stream>>>(stats);
        add_stats<<<dim3(32, 8), 256, 0, stream>>>(net_f, tmp, stats, 0);
        bn_norm<<<dim3(32, 8), 256, 0, stream>>>(net_f, net_b, stats, gamma1, beta1, t);

        // net = BN(net + relu(net @ Wd))
        gemm_bf16_wmma<<<dim3(4, 32, 8), 256, 0, stream>>>(
            net_b, NH, Wd_b + (long)t * 256 * 256, 4L * 256 * 256,
            tmp, nullptr, NH, 256, 256, 256, 256);
        zero_stats<<<16, 256, 0, stream>>>(stats);
        add_stats<<<dim3(32, 8), 256, 0, stream>>>(net_f, tmp, stats, 1);
        bn_norm<<<dim3(32, 8), 256, 0, stream>>>(net_f, net_b, stats, gamma2, beta2, t);
    }

    // [H][B*S][V] -> [B,S,H,V]
    write_out<<<dim3(4096, 8), 256, 0, stream>>>(net_f, (float*)d_out);
}
